// Attention_5394478924244
// MI455X (gfx1250) — matrix operand order repaired and checked by reading, not compiled, not run on hardware
//
#include <hip/hip_runtime.h>

// ---------------------------------------------------------------------------
// Shapes (fixed by the reference)
// ---------------------------------------------------------------------------
#define BATCH 256
#define TLEN  256
#define DIM   512
#define HID   512
#define VOC   96
#define NSTEP 26
#define KIN   (DIM + VOC)       // 608
#define G4H   (4 * HID)         // 2048
#define KCAT  (DIM + HID)       // 1024 (context | h)

typedef unsigned short u16;
typedef __attribute__((ext_vector_type(16))) __bf16 v16bf;
typedef __attribute__((ext_vector_type(8)))  float  v8f;

union FragAB { uint4 u[2]; v16bf v; };

__device__ __forceinline__ u16 f2bf(float f) {
  unsigned u = __float_as_uint(f);
  unsigned r = (u + 0x7FFFu + ((u >> 16) & 1u)) >> 16;   // round-to-nearest-even
  return (u16)r;
}
__device__ __forceinline__ float sigmoidf(float x) {
  return 1.0f / (1.0f + __expf(-x));
}

// ---------------------------------------------------------------------------
// bf16 WMMA GEMM:  C[M,N] = A[M,K] * W[N,K]^T (+ bias[N])
//   A row-major bf16 (lda), W row-major bf16 (ldw), C row-major fp32 (ldc).
//   Block = 256 threads = 8 wave32, block tile 128(M) x 128(N);
//   waves arranged 4(M) x 2(N), each wave owns a 32x64 output:
//   8 f32 accumulators -> 8 WMMAs per K-step from 6 fragment gathers
//   (12 ds_load_b128 per 8 v_wmma: 1.5 LDS ops per matrix op).
//   K stepped by 32 through double-buffered LDS with register prefetch:
//   one barrier per K-step, global loads for step k+1 overlap WMMAs of step k.
//   Requirements: M % 128 == 0, K % 32 == 0 (true for all call sites).
//   grid = (ceil(N/128), M/128).
// ---------------------------------------------------------------------------
__global__ __launch_bounds__(256)
void wmma_gemm_bf16(const u16* __restrict__ A, int lda,
                    const u16* __restrict__ W, int ldw,
                    const float* __restrict__ bias,
                    float* __restrict__ C, int ldc,
                    int N, int K)
{
  __shared__ __align__(16) u16 As[2][128 * 32];   // 2 x 8 KB
  __shared__ __align__(16) u16 Bs[2][128 * 32];   // 2 x 8 KB

  const int tid  = threadIdx.x;
  const int lane = tid & 31;
  const int wv   = tid >> 5;          // 0..7
  const int mw   = wv & 3;            // 0..3  (M sub-tile of wave, 32 rows)
  const int nw   = wv >> 2;           // 0..1  (N sub-tile of wave, 64 cols)
  const int M0    = blockIdx.y << 7;
  const int Nbase = blockIdx.x << 7;

  // Cooperative staging: 2 A-chunks + 2 B-chunks (uint4 = 8 bf16) per thread.
  const int row0 = tid >> 2;                 // 0..63
  const int row1 = 64 + (tid >> 2);          // 64..127
  const int col  = (tid & 3) << 3;           // 0,8,16,24

  // Clamp W rows at the N edge (keeps loads in-bounds & EXEC uniform).
  int wr0 = Nbase + row0; if (wr0 > N - 1) wr0 = N - 1;
  int wr1 = Nbase + row1; if (wr1 > N - 1) wr1 = N - 1;

  const u16* aptr0 = A + (size_t)(M0 + row0) * lda + col;
  const u16* aptr1 = A + (size_t)(M0 + row1) * lda + col;
  const u16* bptr0 = W + (size_t)wr0 * ldw + col;
  const u16* bptr1 = W + (size_t)wr1 * ldw + col;

  // Prefetch K-step 0 straight into buffer 0.
  uint4 ra0 = *(const uint4*)(aptr0);
  uint4 ra1 = *(const uint4*)(aptr1);
  uint4 rb0 = *(const uint4*)(bptr0);
  uint4 rb1 = *(const uint4*)(bptr1);
  *(uint4*)&As[0][row0 * 32 + col] = ra0;
  *(uint4*)&As[0][row1 * 32 + col] = ra1;
  *(uint4*)&Bs[0][row0 * 32 + col] = rb0;
  *(uint4*)&Bs[0][row1 * 32 + col] = rb1;

  v8f acc[2][4];
  #pragma unroll
  for (int mt = 0; mt < 2; ++mt)
    #pragma unroll
    for (int nt = 0; nt < 4; ++nt)
      acc[mt][nt] = (v8f){0.f, 0.f, 0.f, 0.f, 0.f, 0.f, 0.f, 0.f};

  const int ksteps = K >> 5;
  for (int ks = 0; ks < ksteps; ++ks) {
    const int cur = ks & 1;
    __syncthreads();                         // buffer[cur] ready for reads

    const bool more = (ks + 1) < ksteps;
    if (more) {                              // issue next tile's global loads
      const int koff = (ks + 1) << 5;
      ra0 = *(const uint4*)(aptr0 + koff);
      ra1 = *(const uint4*)(aptr1 + koff);
      rb0 = *(const uint4*)(bptr0 + koff);
      rb1 = *(const uint4*)(bptr1 + koff);
    }

    // Per-lane fragment gather per ISA 7.12.2 (16-bit A 16x32 / B 32x16):
    //  A: lanes 0-15 -> K0..7 + K16..23 ; lanes 16-31 -> K8..15 + K24..31
    //  B: lane n (0-15) col n K0..7 + K16..23 ; lanes 16-31 col n K8..15 + K24..31
    const char* asb = (const char*)As[cur];
    const char* bsb = (const char*)Bs[cur];
    const int r  = lane & 15;
    const int hi = lane >> 4;
    FragAB fa[2], fb[4];
    #pragma unroll
    for (int mt = 0; mt < 2; ++mt) {
      const int ba = (mw * 32 + mt * 16 + r) * 64 + hi * 16;
      fa[mt].u[0] = *(const uint4*)(asb + ba);
      fa[mt].u[1] = *(const uint4*)(asb + ba + 32);
    }
    #pragma unroll
    for (int nt = 0; nt < 4; ++nt) {
      const int bb = (nw * 64 + nt * 16 + r) * 64 + hi * 16;
      fb[nt].u[0] = *(const uint4*)(bsb + bb);
      fb[nt].u[1] = *(const uint4*)(bsb + bb + 32);
    }
    #pragma unroll
    for (int mt = 0; mt < 2; ++mt)
      #pragma unroll
      for (int nt = 0; nt < 4; ++nt)
        acc[mt][nt] = __builtin_amdgcn_wmma_f32_16x16x32_bf16(
            false, fa[mt].v, false, fb[nt].v, (short)0, acc[mt][nt],
            false, false);

    if (more) {                              // park next tile in other buffer
      const int nxt = cur ^ 1;
      *(uint4*)&As[nxt][row0 * 32 + col] = ra0;
      *(uint4*)&As[nxt][row1 * 32 + col] = ra1;
      *(uint4*)&Bs[nxt][row0 * 32 + col] = rb0;
      *(uint4*)&Bs[nxt][row1 * 32 + col] = rb1;
    }
  }

  // D layout: lane&15 = column n; lanes<16 rows 0..7, lanes>=16 rows 8..15
  const int r  = lane & 15;
  const int mb = (lane >> 4) ? 8 : 0;
  #pragma unroll
  for (int nt = 0; nt < 4; ++nt) {
    const int n = Nbase + nw * 64 + nt * 16 + r;
    if (n < N) {
      const float bv = bias ? bias[n] : 0.f;
      #pragma unroll
      for (int mt = 0; mt < 2; ++mt) {
        const int mrow = M0 + mw * 32 + mt * 16 + mb;
        #pragma unroll
        for (int rr = 0; rr < 8; ++rr)
          C[(size_t)(mrow + rr) * ldc + n] = acc[mt][nt][rr] + bv;
      }
    }
  }
}

// ---------------------------------------------------------------------------
// Setup / conversion kernels
// ---------------------------------------------------------------------------
__global__ void cvt_f32_bf16(const float* __restrict__ s, u16* __restrict__ d, int n) {
  for (int i = blockIdx.x * blockDim.x + threadIdx.x; i < n;
       i += gridDim.x * blockDim.x)
    d[i] = f2bf(s[i]);
}

// Wcomb[n, 0:512] = W_ih[n, 0:512] ; Wcomb[n, 512:1024] = W_hh[n, :]
__global__ void pack_wcomb(const float* __restrict__ Wih,
                           const float* __restrict__ Whh,
                           u16* __restrict__ d) {
  int i = blockIdx.x * blockDim.x + threadIdx.x;   // exactly G4H*KCAT threads
  int nrow = i >> 10, k = i & 1023;
  float v = (k < DIM) ? Wih[(size_t)nrow * KIN + k]
                      : Whh[(size_t)nrow * HID + (k - DIM)];
  d[i] = f2bf(v);
}

__global__ void add_vec(const float* a, const float* b, float* d, int n) {
  int i = blockIdx.x * blockDim.x + threadIdx.x;
  if (i < n) d[i] = a[i] + b[i];
}
__global__ void zero_f32(float* p, int n) {
  int i = blockIdx.x * blockDim.x + threadIdx.x;
  if (i < n) p[i] = 0.f;
}
__global__ void zero_u16(u16* p, int n) {
  int i = blockIdx.x * blockDim.x + threadIdx.x;
  if (i < n) p[i] = 0;
}

// ---------------------------------------------------------------------------
// Fused attention: e = tanh(proj_H[b,t,:] + proj_h[b,:]) . w_score
//                  alpha = softmax_t(e);  ctx[b,:] = sum_t alpha * batch_H[b,t,:]
// Writes ctx as bf16 into xcat[b, 0:DIM].  One block per batch row.
// ---------------------------------------------------------------------------
__global__ __launch_bounds__(256)
void attention_kernel(const float* __restrict__ projH,   // [B,T,H]
                      const float* __restrict__ projh,   // [B,H]
                      const float* __restrict__ wscore,  // [H]
                      const float* __restrict__ batchH,  // [B,T,D]
                      u16* __restrict__ xcat)            // [B, KCAT] bf16
{
  __shared__ float sProj[HID];
  __shared__ float sW[HID];
  __shared__ float sE[TLEN];
  __shared__ float red[256];

  const int b = blockIdx.x;
  const int tid = threadIdx.x;
  const int lane = tid & 31;
  const int wv = tid >> 5;

  for (int j = tid; j < HID; j += 256) {
    sProj[j] = projh[(size_t)b * HID + j];
    sW[j]    = wscore[j];
  }
  __syncthreads();

  // one wave per t; lane-strided reduction over H (coalesced 128B reads)
  for (int t = wv; t < TLEN; t += 8) {
    const float* ph = projH + ((size_t)b * TLEN + t) * HID;
    float sum = 0.f;
    for (int j = lane; j < HID; j += 32)
      sum += tanhf(ph[j] + sProj[j]) * sW[j];
    #pragma unroll
    for (int off = 16; off >= 1; off >>= 1)
      sum += __shfl_down(sum, off, 32);
    if (lane == 0) sE[t] = sum;
  }
  __syncthreads();

  // softmax over T=256 (one element per thread)
  float e = sE[tid];
  red[tid] = e; __syncthreads();
  #pragma unroll
  for (int s = 128; s > 0; s >>= 1) {
    if (tid < s) red[tid] = fmaxf(red[tid], red[tid + s]);
    __syncthreads();
  }
  float m = red[0]; __syncthreads();
  float ex = __expf(e - m);
  red[tid] = ex; __syncthreads();
  #pragma unroll
  for (int s = 128; s > 0; s >>= 1) {
    if (tid < s) red[tid] += red[tid + s];
    __syncthreads();
  }
  float alpha = ex / red[0];
  __syncthreads();
  sE[tid] = alpha;
  __syncthreads();

  // context: ctx[d] = sum_t alpha[t] * batch_H[b,t,d]
  for (int d = tid; d < DIM; d += 256) {
    float acc = 0.f;
    const float* bh = batchH + (size_t)b * TLEN * DIM + d;
    for (int t = 0; t < TLEN; ++t)
      acc += sE[t] * bh[(size_t)t * DIM];
    xcat[(size_t)b * KCAT + d] = f2bf(acc);
  }
}

// ---------------------------------------------------------------------------
// LSTM pointwise: adds the exact fp32 one-hot column of W_ih, applies gates,
// updates c, writes new h (bf16) into xcat[b, DIM:KCAT].
// ---------------------------------------------------------------------------
__global__ __launch_bounds__(256)
void lstm_pointwise(const float* __restrict__ gates,   // [B,4H]
                    const float* __restrict__ Wih,     // fp32 [4H, KIN]
                    const int* __restrict__ text,      // [B, NSTEP]
                    int step,
                    float* __restrict__ c,             // [B,H]
                    u16* __restrict__ xcat)            // [B,KCAT]
{
  int idx = blockIdx.x * blockDim.x + threadIdx.x;     // B*H threads
  int b = idx >> 9, j = idx & 511;
  int tok = text[b * NSTEP + step];
  size_t base = (size_t)b * G4H;
  float gi = gates[base + j]            + Wih[(size_t)(j)          * KIN + DIM + tok];
  float gf = gates[base + HID + j]      + Wih[(size_t)(HID + j)    * KIN + DIM + tok];
  float gg = gates[base + 2 * HID + j]  + Wih[(size_t)(2*HID + j)  * KIN + DIM + tok];
  float go = gates[base + 3 * HID + j]  + Wih[(size_t)(3*HID + j)  * KIN + DIM + tok];
  float cp = c[idx];
  float cn = sigmoidf(gf) * cp + sigmoidf(gi) * tanhf(gg);
  float hn = sigmoidf(go) * tanhf(cn);
  c[idx] = cn;
  xcat[(size_t)b * KCAT + DIM + j] = f2bf(hn);
}

// ---------------------------------------------------------------------------
// Host orchestration
// ---------------------------------------------------------------------------
extern "C" void kernel_launch(void* const* d_in, const int* in_sizes, int n_in,
                              void* d_out, int out_size, void* d_ws, size_t ws_size,
                              hipStream_t stream) {
  const float* batch_H = (const float*)d_in[0];
  const float* W_i2h   = (const float*)d_in[1];
  const float* W_h2h   = (const float*)d_in[2];
  const float* b_h2h   = (const float*)d_in[3];
  const float* w_score = (const float*)d_in[4];
  const float* W_ih    = (const float*)d_in[5];
  const float* W_hh    = (const float*)d_in[6];
  const float* b_ih    = (const float*)d_in[7];
  const float* b_hh    = (const float*)d_in[8];
  const float* W_gen   = (const float*)d_in[9];
  const float* b_gen   = (const float*)d_in[10];
  const int*   text    = (const int*)d_in[11];
  float* out = (float*)d_out;                 // [B, NSTEP, V]

  char* ws = (char*)d_ws;
  size_t o = 0;
  auto take = [&](size_t bytes) { void* p = ws + o; o = (o + bytes + 255) & ~(size_t)255; return p; };

  float* projH   = (float*)take((size_t)BATCH * TLEN * HID * 4);   // 134.2 MB
  u16*   batchHb = (u16*)  take((size_t)BATCH * TLEN * DIM * 2);   // 67.1 MB
  u16*   Wcomb   = (u16*)  take((size_t)G4H * KCAT * 2);           //  4.2 MB
  u16*   Wi2hb   = (u16*)  take((size_t)HID * DIM * 2);
  u16*   Wh2hb   = (u16*)  take((size_t)HID * HID * 2);
  u16*   Wgenb   = (u16*)  take((size_t)VOC * HID * 2);
  float* bcomb   = (float*)take((size_t)G4H * 4);
  float* projh   = (float*)take((size_t)BATCH * HID * 4);
  float* gates   = (float*)take((size_t)BATCH * G4H * 4);
  u16*   xcat    = (u16*)  take((size_t)BATCH * KCAT * 2);
  float* cstate  = (float*)take((size_t)BATCH * HID * 4);
  (void)ws_size; (void)in_sizes; (void)n_in; (void)out_size;

  // --- one-time conversions -------------------------------------------------
  {
    int n = BATCH * TLEN * DIM;
    cvt_f32_bf16<<<4096, 256, 0, stream>>>(batch_H, batchHb, n);
    cvt_f32_bf16<<<(HID * DIM + 255) / 256, 256, 0, stream>>>(W_i2h, Wi2hb, HID * DIM);
    cvt_f32_bf16<<<(HID * HID + 255) / 256, 256, 0, stream>>>(W_h2h, Wh2hb, HID * HID);
    cvt_f32_bf16<<<(VOC * HID + 255) / 256, 256, 0, stream>>>(W_gen, Wgenb, VOC * HID);
    pack_wcomb<<<(G4H * KCAT) / 256, 256, 0, stream>>>(W_ih, W_hh, Wcomb);
    add_vec<<<G4H / 256, 256, 0, stream>>>(b_ih, b_hh, bcomb, G4H);
    zero_f32<<<(BATCH * HID) / 256, 256, 0, stream>>>(cstate, BATCH * HID);
    zero_u16<<<(BATCH * KCAT) / 256, 256, 0, stream>>>(xcat, BATCH * KCAT);
  }

  // --- proj_H = batch_H @ W_i2h^T   [65536, 512] ---------------------------
  {
    dim3 grid(HID / 128, (BATCH * TLEN) / 128);
    wmma_gemm_bf16<<<grid, 256, 0, stream>>>(batchHb, DIM, Wi2hb, DIM,
                                             nullptr, projH, HID, HID, DIM);
  }

  // --- decode loop ----------------------------------------------------------
  for (int s = 0; s < NSTEP; ++s) {
    // proj_h = h @ W_h2h^T + b_h2h   (h = xcat[:, DIM:] bf16, lda = KCAT)
    {
      dim3 grid(HID / 128, BATCH / 128);
      wmma_gemm_bf16<<<grid, 256, 0, stream>>>(xcat + DIM, KCAT, Wh2hb, HID,
                                               b_h2h, projh, HID, HID, HID);
    }
    // attention -> xcat[:, 0:DIM] = bf16(context)
    attention_kernel<<<BATCH, 256, 0, stream>>>(projH, projh, w_score,
                                                batch_H, xcat);
    // gates = [ctx|h] @ [W_ih[:, :D] | W_hh]^T + (b_ih + b_hh)
    {
      dim3 grid(G4H / 128, BATCH / 128);
      wmma_gemm_bf16<<<grid, 256, 0, stream>>>(xcat, KCAT, Wcomb, KCAT,
                                               bcomb, gates, G4H, G4H, KCAT);
    }
    // LSTM pointwise (+ exact one-hot column), update c, write new h
    lstm_pointwise<<<(BATCH * HID) / 256, 256, 0, stream>>>(gates, W_ih, text,
                                                            s, cstate, xcat);
    // logits = h_new @ W_gen^T + b_gen -> out[:, s, :] (strided C)
    {
      dim3 grid((VOC + 127) / 128, BATCH / 128);
      wmma_gemm_bf16<<<grid, 256, 0, stream>>>(xcat + DIM, KCAT, Wgenb, HID,
                                               b_gen, out + (size_t)s * VOC,
                                               NSTEP * VOC, VOC, HID);
    }
  }
}
